// KeyframeSelectionNetwork_70660801954363
// MI455X (gfx1250) — matrix operand order repaired and checked by reading, not compiled, not run on hardware
//
#include <hip/hip_runtime.h>
#include <hip/hip_bf16.h>
#include <stdint.h>

typedef __attribute__((ext_vector_type(16))) __bf16 v16bf;
typedef __attribute__((ext_vector_type(8)))  float  v8f;

#define F_DIM 1024
#define T_DIM 64
#define NBV   512        // B*V
#define HID   256
#define KMLP  8192       // V*F
#define BATCH 64

// workspace layout (bytes)
#define WT_OFF    (0u)                  // bf16[1024*1024]  W_gcn^T   (2 MB)
#define W1T_OFF   (2u * 1024 * 1024)    // bf16[256*8192]   W1^T      (4 MB)
#define POOL_OFF  (6u * 1024 * 1024)    // float[512*1024]  pooled    (2 MB)
#define H_OFF     (8u * 1024 * 1024)    // float[64*256]    hidden    (64 KB)

// ---------------------------------------------------------------------------
// helpers: CDNA5 async copy global->LDS (ASYNCcnt-tracked) + barriers
// ---------------------------------------------------------------------------
__device__ __forceinline__ uint32_t lds_u32(const void* p) {
  // shared-aperture flat pointers keep the LDS byte offset in the low 32 bits
  return (uint32_t)(uintptr_t)p;
}

__device__ __forceinline__ void async_copy16(uint32_t lds_addr, const void* gaddr) {
  asm volatile("global_load_async_to_lds_b128 %0, %1, off"
               :: "v"(lds_addr), "v"(gaddr) : "memory");
}

#define WAIT_ASYNCCNT(n) asm volatile("s_wait_asynccnt %0" :: "i"(n) : "memory")

// Workgroup barrier WITHOUT the implicit vmem-drain of __syncthreads():
// used only to fence LDS-buffer reuse (ds reads are already drained by the
// per-WMMA dscnt waits), so in-flight A prefetch loads may stay outstanding.
__device__ __forceinline__ void wg_barrier_relaxed() {
  asm volatile("s_barrier_signal -1\n\ts_barrier_wait -1" ::: "memory");
}

// Stage one B tile: 128 cols x 32 k of bf16 = 8192 bytes, 128 threads x 4 x b128.
// LDS layout: [col][k] bf16, col stride 64 bytes.
__device__ __forceinline__ void issue_b_stage(const __bf16* wt_tile, uint32_t lds_base,
                                              int tid) {
#pragma unroll
  for (int i = 0; i < 4; ++i) {
    const int o    = tid * 16 + i * 2048;   // byte offset in tile
    const int row  = o >> 6;                // 0..127 (weight column)
    const int byte = o & 63;
    async_copy16(lds_base + o, wt_tile + (size_t)row * F_DIM + (byte >> 1));
  }
}

// ---------------------------------------------------------------------------
// Kernel 0: f32 [K,N] row-major -> bf16 [N,K] row-major (transpose+convert)
// ---------------------------------------------------------------------------
__global__ __launch_bounds__(256) void cvt_transpose_kernel(
    const float* __restrict__ src, __bf16* __restrict__ dst, int K, int N) {
  long idx = (long)blockIdx.x * 256 + threadIdx.x;
  if (idx >= (long)K * N) return;
  int n = (int)(idx / K);
  int k = (int)(idx - (long)n * K);
  dst[idx] = (__bf16)src[(size_t)k * N + n];
}

// ---------------------------------------------------------------------------
// Kernel 1: fused  x = videos @ W_gcn  (bf16 WMMA, f32 acc)
//           + GCN symmetric-norm fix for frame t==1
//           + max-pool over the 64 frames + b_gcn
// grid = (512 streams, 8 feature blocks of 128), block = 128 (4 waves)
// B tile double-buffered in LDS via async-to-LDS DMA; A double-buffered in regs.
// Steady-state iteration is branch-free; final k-step peeled.
// ---------------------------------------------------------------------------
struct GcnCtx {
  v8f acc[8];
  v8f fa, fb;
};

__device__ __forceinline__ void gcn_compute_step(GcnCtx& c, const __bf16* sbuf,
                                                 int l16, int hi) {
  // convert current A fragment: lane = row M=l16, K=(hi*16..+15)
  v16bf a;
#pragma unroll
  for (int i = 0; i < 8; ++i) { a[i] = (__bf16)c.fa[i]; a[8 + i] = (__bf16)c.fb[i]; }

  // batch B-fragment LDS reads, then WMMAs (scheduler rolls them together)
  const __bf16* sb = sbuf + hi * 16;
  v16bf b[8];
#pragma unroll
  for (int nt = 0; nt < 8; ++nt)
    b[nt] = *(const v16bf*)(sb + (nt * 16 + l16) * 32);
#pragma unroll
  for (int nt = 0; nt < 8; ++nt)
    c.acc[nt] = __builtin_amdgcn_wmma_f32_16x16x32_bf16(false, a, false, b[nt],
                                                        (short)0, c.acc[nt], false, false);
}

__global__ __launch_bounds__(128) void gcn_gemm_pool_kernel(
    const float* __restrict__ videos, const __bf16* __restrict__ Wt,
    const float* __restrict__ b_gcn, float* __restrict__ pooled) {
  __shared__ __align__(16) __bf16 smem[2][128 * 32];  // 2 x 8 KB B stages
  __shared__ float redbuf[8 * 128];                   // pooling scratch

  const int tid  = threadIdx.x;
  const int wave = tid >> 5;        // 0..3  -> frames 16w..16w+15
  const int lane = tid & 31;
  const int l16  = lane & 15;
  const int hi   = lane >> 4;
  const int bv   = blockIdx.x;        // video stream (b*V + v)
  const int n0   = blockIdx.y * 128;  // feature block

  const float*  aptr    = videos + (size_t)(bv * T_DIM + wave * 16 + l16) * F_DIM;
  const __bf16* wt_base = Wt + (size_t)n0 * F_DIM;
  const uint32_t lb0 = lds_u32(&smem[0][0]);
  const uint32_t lb1 = lds_u32(&smem[1][0]);

  GcnCtx c;
#pragma unroll
  for (int nt = 0; nt < 8; ++nt) c.acc[nt] = v8f{};

  // prologue: DMA stage 0, preload A fragment for k0=0
  issue_b_stage(wt_base, lb0, tid);
  c.fa = *(const v8f*)(aptr + hi * 16);
  c.fb = *(const v8f*)(aptr + 8 + hi * 16);

  // steady state: branch-free (final k-step peeled below)
  for (int k0 = 0; k0 < F_DIM - 32; k0 += 32) {
    const int buf = (k0 >> 5) & 1;
    issue_b_stage(wt_base + k0 + 32, buf ? lb0 : lb1, tid);  // DMA next stage
    WAIT_ASYNCCNT(4);     // current stage landed; next 4 ops stay in flight
    __syncthreads();      // current B stage visible to all waves

    // issue next A loads early (overlap with WMMA below; drain next iter)
    v8f na = *(const v8f*)(aptr + k0 + 32 + hi * 16);
    v8f nb = *(const v8f*)(aptr + k0 + 40 + hi * 16);

    gcn_compute_step(c, &smem[buf][0], l16, hi);

    c.fa = na; c.fb = nb;
    wg_barrier_relaxed(); // fence buffer reuse only; A loads stay in flight
  }

  // peeled final k-step
  WAIT_ASYNCCNT(0);
  __syncthreads();
  gcn_compute_step(c, &smem[((F_DIM >> 5) - 1) & 1][0], l16, hi);

  // GCN fix: only frame 1 differs: out1 = 0.5*x1 + (1/sqrt2)*x0.
  // C layout: VGPR r = row M=r for lanes 0-15 (hi==0); rows 0,1 live in wave 0.
  if (wave == 0 && hi == 0) {
#pragma unroll
    for (int nt = 0; nt < 8; ++nt)
      c.acc[nt][1] = c.acc[nt][1] * 0.5f + c.acc[nt][0] * 0.70710678118654752f;
  }

  // max over the 8 rows this lane holds, per N-tile -> LDS
  __syncthreads();
#pragma unroll
  for (int nt = 0; nt < 8; ++nt) {
    float m = c.acc[nt][0];
#pragma unroll
    for (int r = 1; r < 8; ++r) m = fmaxf(m, c.acc[nt][r]);
    redbuf[(wave * 2 + hi) * 128 + nt * 16 + l16] = m;
  }
  __syncthreads();

  {
    float v = redbuf[tid];
#pragma unroll
    for (int i = 1; i < 8; ++i) v = fmaxf(v, redbuf[i * 128 + tid]);
    pooled[(size_t)bv * F_DIM + n0 + tid] = v + b_gcn[n0 + tid];
  }
}

// ---------------------------------------------------------------------------
// Kernel 2: h = relu(pooled[64,8192] @ W1 + b1)   (bf16 WMMA, f32 acc)
// grid = 4 blocks (N tiles of 64), block = 128 (4 waves over M=64)
// ---------------------------------------------------------------------------
__global__ __launch_bounds__(128) void mlp1_kernel(
    const float* __restrict__ pooled, const __bf16* __restrict__ W1t,
    const float* __restrict__ b1, float* __restrict__ h) {
  const int tid  = threadIdx.x;
  const int wave = tid >> 5;
  const int lane = tid & 31;
  const int l16  = lane & 15;
  const int hi   = lane >> 4;
  const int n0   = blockIdx.x * 64;

  const float*  aptr = pooled + (size_t)(wave * 16 + l16) * KMLP;
  const __bf16* bptr = W1t + (size_t)(n0 + l16) * KMLP;

  v8f acc[4] = {v8f{}, v8f{}, v8f{}, v8f{}};

  for (int k0 = 0; k0 < KMLP; k0 += 32) {
    const float* ap = aptr + k0 + hi * 16;
    v8f f1 = *(const v8f*)(ap);
    v8f f2 = *(const v8f*)(ap + 8);

    // batch the 4 B-fragment loads, then convert A, then 4 WMMAs
    const __bf16* bp = bptr + k0 + hi * 16;
    v16bf b[4];
#pragma unroll
    for (int nt = 0; nt < 4; ++nt)
      b[nt] = *(const v16bf*)(bp + nt * 16 * KMLP);

    v16bf a;
#pragma unroll
    for (int i = 0; i < 8; ++i) { a[i] = (__bf16)f1[i]; a[8 + i] = (__bf16)f2[i]; }

#pragma unroll
    for (int nt = 0; nt < 4; ++nt)
      acc[nt] = __builtin_amdgcn_wmma_f32_16x16x32_bf16(false, a, false, b[nt],
                                                        (short)0, acc[nt], false, false);
  }

#pragma unroll
  for (int nt = 0; nt < 4; ++nt) {
    const int n = n0 + nt * 16 + l16;
    const float bias = b1[n];
#pragma unroll
    for (int r = 0; r < 8; ++r) {
      const int m = wave * 16 + hi * 8 + r;
      h[m * HID + n] = fmaxf(acc[nt][r] + bias, 0.0f);
    }
  }
}

// ---------------------------------------------------------------------------
// Kernel 3: out = sigmoid(h[64,256] @ W2[256,512] + b2)  (tiny; plain VALU)
// ---------------------------------------------------------------------------
__global__ __launch_bounds__(256) void mlp2_sigmoid_kernel(
    const float* __restrict__ h, const float* __restrict__ W2,
    const float* __restrict__ b2, float* __restrict__ out) {
  const int idx = blockIdx.x * 256 + threadIdx.x;   // 0..32767
  const int b = idx >> 9;                           // batch
  const int j = idx & 511;                          // v*T + t
  const float* hp = h + b * HID;
  float s = b2[j];
#pragma unroll 4
  for (int k = 0; k < HID; ++k) s = fmaf(hp[k], W2[k * 512 + j], s);
  out[idx] = 1.0f / (1.0f + __expf(-s));
}

// ---------------------------------------------------------------------------
extern "C" void kernel_launch(void* const* d_in, const int* in_sizes, int n_in,
                              void* d_out, int out_size, void* d_ws, size_t ws_size,
                              hipStream_t stream) {
  const float* videos = (const float*)d_in[0];   // [64,8,64,1024]
  const float* W_gcn  = (const float*)d_in[1];   // [1024,1024]
  const float* b_gcn  = (const float*)d_in[2];   // [1024]
  const float* W1     = (const float*)d_in[3];   // [8192,256]
  const float* b1     = (const float*)d_in[4];   // [256]
  const float* W2     = (const float*)d_in[5];   // [256,512]
  const float* b2     = (const float*)d_in[6];   // [512]
  float* out = (float*)d_out;                    // [64,8,64] flat

  char* ws = (char*)d_ws;
  __bf16* Wt     = (__bf16*)(ws + WT_OFF);
  __bf16* W1t    = (__bf16*)(ws + W1T_OFF);
  float*  pooled = (float*)(ws + POOL_OFF);
  float*  h      = (float*)(ws + H_OFF);

  // weight convert+transpose to bf16
  cvt_transpose_kernel<<<(1024 * 1024) / 256, 256, 0, stream>>>(W_gcn, Wt, 1024, 1024);
  cvt_transpose_kernel<<<(8192 * 256) / 256, 256, 0, stream>>>(W1, W1t, 8192, 256);

  // fused GCN GEMM + norm + max-pool (dominant cost; WMMA + async-LDS pipeline)
  gcn_gemm_pool_kernel<<<dim3(NBV, F_DIM / 128), 128, 0, stream>>>(videos, Wt, b_gcn, pooled);

  // MLP layer 1 (WMMA) and layer 2 (sigmoid)
  mlp1_kernel<<<HID / 64, 128, 0, stream>>>(pooled, W1t, b1, h);
  mlp2_sigmoid_kernel<<<(BATCH * 512) / 256, 256, 0, stream>>>(h, W2, b2, out);
}
